// Span2oScorer_7730941132800
// MI455X (gfx1250) — compile-verified, matrix operand order checked by reading, not run on hardware
//
#include <hip/hip_runtime.h>
#include <hip/hip_bf16.h>

typedef __attribute__((ext_vector_type(16))) _Float16 v16h;
typedef __attribute__((ext_vector_type(8)))  float    v8f;

#define WMMA_F16(a, b, c) \
  __builtin_amdgcn_wmma_f32_16x16x32_f16(false, (a), false, (b), (short)0, (c), false, false)

// ---- Tile loader ------------------------------------------------------------
// A/B operand for v_wmma_f32_16x16x32_f16 (ISA 7.12.2, 16-bit 16x32 layout):
// lane L: row = L&15 (M for A, N for B-transposed), h = L>>4.
// halves [0..7]  = K (h*8 .. h*8+7), halves [8..15] = K (16+h*8 .. 16+h*8+7)
// => two contiguous 16-byte runs -> two b128 loads.
__device__ __forceinline__ v16h load_tile_f16(const _Float16* __restrict__ base,
                                              int ld, int row0, int k0, int lane) {
  int r = lane & 15, h = lane >> 4;
  const _Float16* p = base + (size_t)(row0 + r) * ld + k0 + h * 8;
  union { v16h v; uint4 q[2]; } u;
  u.q[0] = *(const uint4*)(p);
  u.q[1] = *(const uint4*)(p + 16);
  return u.v;
}

// ---- flat f32 -> f16 convert ------------------------------------------------
__global__ void k_cvt(const float* __restrict__ src, _Float16* __restrict__ dst, int n) {
  int i = blockIdx.x * blockDim.x + threadIdx.x;
  if (i < n) dst[i] = (_Float16)src[i];
}

// ---- W3 (129,128,129) f32  ->  W3T[(j*160+i)*128 + k] f16, zero-padded ------
__global__ void k_pack_w3(const float* __restrict__ W3, _Float16* __restrict__ W3T) {
  int idx = blockIdx.x * blockDim.x + threadIdx.x;
  if (idx >= 25600 * 128) return;
  int k = idx & 127, n = idx >> 7;
  int j = n / 160, i = n - j * 160;
  float v = 0.f;
  if (i < 129 && j < 129) v = W3[((size_t)i * 128 + k) * 129 + j];
  W3T[(size_t)n * 128 + k] = (_Float16)v;
}

// ---- pad columns 128..159 of xq/yq: col128 = 1.0 (bias), rest 0 -------------
__global__ void k_fill_pad(_Float16* __restrict__ xq, _Float16* __restrict__ yq) {
  int idx = blockIdx.x * blockDim.x + threadIdx.x;
  if (idx >= 2048 * 32 * 2) return;
  _Float16* buf = (idx < 2048 * 32) ? xq : yq;
  int t = idx & (2048 * 32 - 1);
  int row = t >> 5, c = 128 + (t & 31);
  buf[(size_t)row * 160 + c] = (_Float16)((c == 128) ? 1.0f : 0.0f);
}

// ---- MLP: dst = leaky_relu(A(2048x1024) @ W(128x1024)^T + bias), f16 --------
// One wave = 16x64 strip; ping-pong double buffering (no register copies).
__global__ void k_mlp(const _Float16* __restrict__ A, const _Float16* __restrict__ W,
                      const float* __restrict__ bias, _Float16* __restrict__ dst,
                      int dstld) {
  int lane = threadIdx.x & 31, wave = threadIdx.x >> 5;
  int t = blockIdx.x * 8 + wave;          // 256 strips = 128 m-tiles x 2 n-groups
  int tm = t >> 1, g = t & 1;
  int m0 = tm * 16, n0 = g * 64;
  v8f acc[4] = {};
  v16h a0, a1;
  v16h b0[4], b1[4];
  a0 = load_tile_f16(A, 1024, m0, 0, lane);
#pragma unroll
  for (int u = 0; u < 4; ++u) b0[u] = load_tile_f16(W, 1024, n0 + u * 16, 0, lane);
  for (int k0 = 0; k0 < 1024; k0 += 64) {
    // prefetch odd step (k0+32 always < 1024 here since steps are even-count)
    a1 = load_tile_f16(A, 1024, m0, k0 + 32, lane);
#pragma unroll
    for (int u = 0; u < 4; ++u)
      b1[u] = load_tile_f16(W, 1024, n0 + u * 16, k0 + 32, lane);
#pragma unroll
    for (int u = 0; u < 4; ++u) acc[u] = WMMA_F16(a0, b0[u], acc[u]);
    if (k0 + 64 < 1024) {
      a0 = load_tile_f16(A, 1024, m0, k0 + 64, lane);
#pragma unroll
      for (int u = 0; u < 4; ++u)
        b0[u] = load_tile_f16(W, 1024, n0 + u * 16, k0 + 64, lane);
    }
#pragma unroll
    for (int u = 0; u < 4; ++u) acc[u] = WMMA_F16(a1, b1[u], acc[u]);
  }
  int r = lane & 15, h = lane >> 4;
#pragma unroll
  for (int u = 0; u < 4; ++u) {
    float bv = bias[n0 + u * 16 + r];
#pragma unroll
    for (int v = 0; v < 8; ++v) {
      float xv = acc[u][v] + bv;
      xv = (xv > 0.f) ? xv : 0.1f * xv;
      dst[(size_t)(m0 + v + 8 * h) * dstld + n0 + u * 16 + r] = (_Float16)xv;
    }
  }
}

// ---- Stage A (per b): wB[z][j*160+i] = sum_k zq[z,k] * W3[i,k,j] ------------
// One wave = 16x64 strip; K=128, fully unrolled ping-pong (4 K-steps).
__global__ void k_stage_a(const _Float16* __restrict__ zq, const _Float16* __restrict__ W3T,
                          _Float16* __restrict__ wB) {
  int lane = threadIdx.x & 31, wave = threadIdx.x >> 5;
  int t = blockIdx.x * 8 + wave;          // 3200 strips = 8 z-tiles x 400 n-groups
  int tm = t & 7, g = t >> 3;
  int m0 = tm * 16, n0 = g * 64;
  v8f acc[4] = {};
  v16h a[2];
  v16h bt[2][4];
  a[0] = load_tile_f16(zq, 128, m0, 0, lane);
#pragma unroll
  for (int u = 0; u < 4; ++u) bt[0][u] = load_tile_f16(W3T, 128, n0 + u * 16, 0, lane);
#pragma unroll
  for (int s = 0; s < 4; ++s) {
    int cur = s & 1, nxt = cur ^ 1;
    if (s < 3) {
      int kn = (s + 1) * 32;
      a[nxt] = load_tile_f16(zq, 128, m0, kn, lane);
#pragma unroll
      for (int u = 0; u < 4; ++u)
        bt[nxt][u] = load_tile_f16(W3T, 128, n0 + u * 16, kn, lane);
    }
#pragma unroll
    for (int u = 0; u < 4; ++u) acc[u] = WMMA_F16(a[cur], bt[cur][u], acc[u]);
  }
  int r = lane & 15, h = lane >> 4;
#pragma unroll
  for (int u = 0; u < 4; ++u)
#pragma unroll
    for (int v = 0; v < 8; ++v)
      wB[(size_t)(m0 + v + 8 * h) * 25600 + n0 + u * 16 + r] = (_Float16)acc[u][v];
}

// ---- Stage B (per b): m[x][z][j] = sum_i x1[x,i] * wB[z][j][i] --------------
// One wave = 16x80 strip (5 j-tiles); K=160, fully unrolled ping-pong.
__global__ void k_stage_b(const _Float16* __restrict__ xq, const _Float16* __restrict__ wB,
                          _Float16* __restrict__ mbuf) {
  int lane = threadIdx.x & 31, wave = threadIdx.x >> 5;
  int t = blockIdx.x * 8 + wave;          // 2048 strips = 128 z x (8 x-tiles x 2 j-groups)
  int z = t >> 4, rem = t & 15;
  int tx = rem >> 1, g = rem & 1;
  int m0 = tx * 16, n0 = g * 80;
  const _Float16* Bz = wB + (size_t)z * 25600;
  v8f acc[5] = {};
  v16h a[2];
  v16h bt[2][5];
  a[0] = load_tile_f16(xq, 160, m0, 0, lane);
#pragma unroll
  for (int u = 0; u < 5; ++u) bt[0][u] = load_tile_f16(Bz, 160, n0 + u * 16, 0, lane);
#pragma unroll
  for (int s = 0; s < 5; ++s) {
    int cur = s & 1, nxt = cur ^ 1;
    if (s < 4) {
      int kn = (s + 1) * 32;
      a[nxt] = load_tile_f16(xq, 160, m0, kn, lane);
#pragma unroll
      for (int u = 0; u < 5; ++u)
        bt[nxt][u] = load_tile_f16(Bz, 160, n0 + u * 16, kn, lane);
    }
#pragma unroll
    for (int u = 0; u < 5; ++u) acc[u] = WMMA_F16(a[cur], bt[cur][u], acc[u]);
  }
  int r = lane & 15, h = lane >> 4;
#pragma unroll
  for (int u = 0; u < 5; ++u)
#pragma unroll
    for (int v = 0; v < 8; ++v)
      mbuf[((size_t)(m0 + v + 8 * h) * 128 + z) * 160 + n0 + u * 16 + r] =
          (_Float16)acc[u][v];
}

// ---- Stage C (per b): out[b,x,y,z] = sum_j y1[y,j] * m[x][z][j] -------------
// One wave = 16x64 strip (4 z-tiles); K=160, fully unrolled ping-pong.
__global__ void k_stage_c(const _Float16* __restrict__ yq, const _Float16* __restrict__ mbuf,
                          float* __restrict__ out) {
  int lane = threadIdx.x & 31, wave = threadIdx.x >> 5;
  int t = blockIdx.x * 8 + wave;          // 2048 strips = 128 x * (8 y-tiles x 2 z-groups)
  int x = t >> 4, rem = t & 15;
  int ty = rem >> 1, g = rem & 1;
  int m0 = ty * 16, n0 = g * 64;
  const _Float16* Bx = mbuf + (size_t)x * 128 * 160;
  v8f acc[4] = {};
  v16h a[2];
  v16h bt[2][4];
  a[0] = load_tile_f16(yq, 160, m0, 0, lane);
#pragma unroll
  for (int u = 0; u < 4; ++u) bt[0][u] = load_tile_f16(Bx, 160, n0 + u * 16, 0, lane);
#pragma unroll
  for (int s = 0; s < 5; ++s) {
    int cur = s & 1, nxt = cur ^ 1;
    if (s < 4) {
      int kn = (s + 1) * 32;
      a[nxt] = load_tile_f16(yq, 160, m0, kn, lane);
#pragma unroll
      for (int u = 0; u < 4; ++u)
        bt[nxt][u] = load_tile_f16(Bx, 160, n0 + u * 16, kn, lane);
    }
#pragma unroll
    for (int u = 0; u < 4; ++u) acc[u] = WMMA_F16(a[cur], bt[cur][u], acc[u]);
  }
  int r = lane & 15, h = lane >> 4;
  float* o = out + (size_t)x * 128 * 128;
#pragma unroll
  for (int u = 0; u < 4; ++u)
#pragma unroll
    for (int v = 0; v < 8; ++v)
      o[(size_t)(m0 + v + 8 * h) * 128 + n0 + u * 16 + r] = acc[u][v]; // z-contiguous
}

extern "C" void kernel_launch(void* const* d_in, const int* in_sizes, int n_in,
                              void* d_out, int out_size, void* d_ws, size_t ws_size,
                              hipStream_t stream) {
  (void)in_sizes; (void)n_in; (void)out_size; (void)ws_size;
  const float* x  = (const float*)d_in[0];
  const float* xs = (const float*)d_in[1];
  const float* Wx = (const float*)d_in[2];
  const float* bx = (const float*)d_in[3];
  const float* Wy = (const float*)d_in[4];
  const float* by = (const float*)d_in[5];
  const float* Wz = (const float*)d_in[6];
  const float* bz = (const float*)d_in[7];
  const float* W3 = (const float*)d_in[8];
  float* out = (float*)d_out;

  // workspace layout (bytes), ~28 MB total, 16B aligned, L2-resident (192 MB L2)
  char* ws = (char*)d_ws;
  _Float16* W3T = (_Float16*)(ws + 0);          // 25600*128*2   = 6,553,600
  _Float16* xq  = (_Float16*)(ws + 6553600);    // 2048*160*2    =   655,360
  _Float16* yq  = (_Float16*)(ws + 7208960);    // 2048*160*2    =   655,360
  _Float16* zq  = (_Float16*)(ws + 7864320);    // 2048*128*2    =   524,288
  _Float16* wB  = (_Float16*)(ws + 8388608);    // 128*25600*2   = 6,553,600
  _Float16* mb  = (_Float16*)(ws + 14942208);   // 128*128*160*2 = 5,242,880
  _Float16* xh  = (_Float16*)(ws + 20185088);   // 2048*1024*2   = 4,194,304
  _Float16* xsh = (_Float16*)(ws + 24379392);   // 2048*1024*2   = 4,194,304
  _Float16* Wxh = (_Float16*)(ws + 28573696);   // 128*1024*2    =   262,144
  _Float16* Wyh = (_Float16*)(ws + 28835840);   // 128*1024*2    =   262,144
  _Float16* Wzh = (_Float16*)(ws + 29097984);   // 128*1024*2    =   262,144

  const int NA = 2048 * 1024, NW = 128 * 1024;
  k_cvt<<<(NA + 255) / 256, 256, 0, stream>>>(x,  xh,  NA);
  k_cvt<<<(NA + 255) / 256, 256, 0, stream>>>(xs, xsh, NA);
  k_cvt<<<(NW + 255) / 256, 256, 0, stream>>>(Wx, Wxh, NW);
  k_cvt<<<(NW + 255) / 256, 256, 0, stream>>>(Wy, Wyh, NW);
  k_cvt<<<(NW + 255) / 256, 256, 0, stream>>>(Wz, Wzh, NW);
  k_pack_w3<<<(25600 * 128 + 255) / 256, 256, 0, stream>>>(W3, W3T);
  k_fill_pad<<<(2048 * 32 * 2 + 255) / 256, 256, 0, stream>>>(xq, yq);

  k_mlp<<<32, 256, 0, stream>>>(xsh, Wxh, bx, xq, 160);
  k_mlp<<<32, 256, 0, stream>>>(xsh, Wyh, by, yq, 160);
  k_mlp<<<32, 256, 0, stream>>>(xh,  Wzh, bz, zq, 128);

  for (int b = 0; b < 16; ++b) {   // serialize batches: reuse wB/mb, stay in L2
    k_stage_a<<<400, 256, 0, stream>>>(zq + b * 128 * 128, W3T, wB);
    k_stage_b<<<256, 256, 0, stream>>>(xq + b * 128 * 160, wB, mb);
    k_stage_c<<<256, 256, 0, stream>>>(yq + b * 128 * 160, mb,
                                       out + (size_t)b * 128 * 128 * 128);
  }
}
